// MMProtGraph_47304769798348
// MI455X (gfx1250) — matrix operand order repaired and checked by reading, not compiled, not run on hardware
//
#include <hip/hip_runtime.h>

// ---------------------------------------------------------------------------
// MMProtGraph on gfx1250: GAT x8 + cross-attention + MLP.
// Dense matmuls -> v_wmma_f32_16x16x32_f16 with fully padded operands
// (branch-free, mask-free fragment loads); irregular edge ops -> coalesced
// per-edge blocks with float-as-ordered-uint atomicMax segment softmax.
// ---------------------------------------------------------------------------

typedef __attribute__((ext_vector_type(16))) _Float16 v16h;
typedef __attribute__((ext_vector_type(8)))  float    v8f;
typedef __attribute__((ext_vector_type(4)))  float    f4;

#define DEVI __device__ __forceinline__

DEVI unsigned f2ord(float f) {
  unsigned u = __float_as_uint(f);
  return (u & 0x80000000u) ? ~u : (u | 0x80000000u);
}
DEVI float ord2f(unsigned u) {
  return (u & 0x80000000u) ? __uint_as_float(u & 0x7fffffffu) : __uint_as_float(~u);
}
DEVI float lrelu_f(float v, float s) { return v > 0.f ? v : v * s; }

// ---------------------------------------------------------------------------
// WMMA GEMM, padded-operand contract (host guarantees):
//   K % 32 == 0 ; A has ceil16(M) readable rows of lda floats (lda % 16 == 0,
//   16B-aligned base) ; B has K rows x ceil16(N) readable cols.
// Loads are unconditional (A: 4x float4, B: 16x strided b32); only the
// epilogue stores are bounds-masked. flags: bit0 bias, bit1 leaky-relu.
// ---------------------------------------------------------------------------
__global__ void gemm_wmma(const float* __restrict__ A, const float* __restrict__ B,
                          const float* __restrict__ bias, float* __restrict__ C,
                          int M, int N, int K, int lda, int ldb, int ldc,
                          int flags, float slope) {
  const int lane = threadIdx.x & 31;
  const int hl   = lane >> 4;      // lane half (0/1)
  const int l15  = lane & 15;
  const int tn   = blockIdx.x * 16;
  const int tm   = blockIdx.y * 16;
  const int arow = tm + l15;
  const int bcol = tn + l15;
  const float* Ap = A + (size_t)arow * lda + (hl << 3);
  const float* Bp = B + (size_t)(hl << 3) * ldb + bcol;
  v8f acc = {};
  for (int k0 = 0; k0 < K; k0 += 32) {
    // A fragment 16x32: element e of lane L holds K = ((e>>3)<<4)+((L>>4)<<3)+(e&7)
    const f4* Ar0 = (const f4*)(Ap + k0);        // K = k0 + hl*8 + 0..7
    const f4* Ar1 = (const f4*)(Ap + k0 + 16);   // K = k0 + 16 + hl*8 + 0..7
    f4 a0 = Ar0[0], a1 = Ar0[1], a2 = Ar1[0], a3 = Ar1[1];
    v16h a, b;
#pragma unroll
    for (int e = 0; e < 4; ++e) {
      a[e]      = (_Float16)a0[e];
      a[e + 4]  = (_Float16)a1[e];
      a[e + 8]  = (_Float16)a2[e];
      a[e + 12] = (_Float16)a3[e];
    }
    // B fragment 32x16: element e of lane L holds K = ((e>>3)<<4)+((L>>4)<<3)+(e&7), col = L&15
    const float* Br = Bp + (size_t)k0 * ldb;
#pragma unroll
    for (int e = 0; e < 16; ++e) b[e] = (_Float16)Br[(size_t)(e < 8 ? e : e + 8) * ldb];
    acc = __builtin_amdgcn_wmma_f32_16x16x32_f16(false, a, false, b,
                                                 (short)0, acc, false, false);
  }
  float bb = ((flags & 1) && bcol < N) ? bias[bcol] : 0.f;
#pragma unroll
  for (int r = 0; r < 8; ++r) {
    int row = tm + r + hl * 8;
    if (row < M && bcol < N) {
      float v = acc[r] + bb;
      if (flags & 2) v = lrelu_f(v, slope);
      C[(size_t)row * ldc + bcol] = v;
    }
  }
}

// zero-padding 2D copy: dst[rowsPad x ldd] = src[rows x cols] (src stride lds)
__global__ void pad2d(const float* __restrict__ src, float* __restrict__ dst,
                      int rows, int cols, int lds, int rowsPad, int ldd) {
  int i = blockIdx.x * blockDim.x + threadIdx.x;
  int total = rowsPad * ldd;
  if (i >= total) return;
  int r = i / ldd, c = i % ldd;
  dst[i] = (r < rows && c < cols) ? src[(size_t)r * lds + c] : 0.f;
}

// -------------------------- small utility kernels --------------------------
__global__ void fill_u32(unsigned* p, unsigned v, int n) {
  int i = blockIdx.x * blockDim.x + threadIdx.x;
  if (i < n) p[i] = v;
}
__global__ void fill_f32(float* p, float v, int n) {
  int i = blockIdx.x * blockDim.x + threadIdx.x;
  if (i < n) p[i] = v;
}
__global__ void lrelu_ip(float* p, int n, float s) {
  int i = blockIdx.x * blockDim.x + threadIdx.x;
  if (i < n) { float v = p[i]; p[i] = lrelu_f(v, s); }
}

// per-node dots with att_src / att_dst (C = 128), LDS tree reduce
__global__ void node_dots(const float* __restrict__ h, const float* __restrict__ as,
                          const float* __restrict__ ad, float* ds_, float* dd_) {
  __shared__ float s1[128], s2[128];
  int n = blockIdx.x, c = threadIdx.x;
  float v = h[(size_t)n * 128 + c];
  s1[c] = v * as[c];
  s2[c] = v * ad[c];
  __syncthreads();
  for (int off = 64; off > 0; off >>= 1) {
    if (c < off) { s1[c] += s1[c + off]; s2[c] += s2[c + off]; }
    __syncthreads();
  }
  if (c == 0) { ds_[n] = s1[0]; dd_[n] = s2[0]; }
}

// we_vec[f] = sum_c We[f,c] * att_edge[c]   (11 x 128 contraction)
__global__ void edge_wvec(const float* __restrict__ We, const float* __restrict__ ae,
                          float* wev) {
  int f = threadIdx.x;
  if (f >= 11) return;
  float s = 0.f;
  for (int c = 0; c < 128; ++c) s += We[f * 128 + c] * ae[c];
  wev[f] = s;
}

// edge logit + running segment max (float atomicMax via ordered-uint)
__global__ void edge_logit(const int* __restrict__ ei, int E,
                           const float* __restrict__ ds_, const float* __restrict__ dd_,
                           const float* __restrict__ eattr, const float* __restrict__ wev,
                           int FE, float* elog, unsigned* nm) {
  int e = blockIdx.x * blockDim.x + threadIdx.x;
  if (e >= E) return;
  int src = ei[e], dst = ei[E + e];
  float a = ds_[src] + dd_[dst];
  if (eattr) {
    float t = 0.f;
    for (int f = 0; f < FE; ++f) t += eattr[(size_t)e * FE + f] * wev[f];
    a += t;
  }
  a = lrelu_f(a, 0.2f);           // GATConv negative_slope
  elog[e] = a;
  atomicMax(&nm[dst], f2ord(a));
}

// exp(a - max) and segment sum
__global__ void edge_exp(const int* __restrict__ ei, int E, float* elog,
                         const unsigned* __restrict__ nm, float* ns) {
  int e = blockIdx.x * blockDim.x + threadIdx.x;
  if (e >= E) return;
  int dst = ei[E + e];
  float x = expf(elog[e] - ord2f(nm[dst]));
  elog[e] = x;
  atomicAdd(&ns[dst], x);
}

// out[dst,:] += h[src,:] * (exp / (sum+eps)); one block (128ch) per edge
__global__ void gat_scatter(const float* __restrict__ h, const float* __restrict__ elog,
                            const float* __restrict__ ns, const int* __restrict__ ei,
                            int E, float* out) {
  int e = blockIdx.x, c = threadIdx.x;
  int src = ei[e], dst = ei[E + e];
  float w = elog[e] / (ns[dst] + 1e-16f);
  atomicAdd(&out[(size_t)dst * 128 + c], h[(size_t)src * 128 + c] * w);
}

// SMILES pooling: 32 contiguous nodes/graph -> lrelu(concat(max, mean))
__global__ void pool_smiles(const float* __restrict__ h, float* pooled) {
  int b = blockIdx.x, c = threadIdx.x;
  float mx = -3.4e38f, sm = 0.f;
  for (int j = 0; j < 32; ++j) {
    float v = h[((size_t)b * 32 + j) * 128 + c];
    mx = fmaxf(mx, v); sm += v;
  }
  float mn = sm * (1.f / 32.f);
  pooled[b * 256 + c]       = lrelu_f(mx, 0.01f);
  pooled[b * 256 + 128 + c] = lrelu_f(mn, 0.01f);
}

// protein pooling: lrelu(segment_max(alpha * h)) over 256 contiguous nodes
__global__ void pool_prot(const float* __restrict__ h, const float* __restrict__ alpha,
                          float* hpool) {
  int b = blockIdx.x, c = threadIdx.x;
  float mx = -3.4e38f;
  for (int j = 0; j < 256; ++j) {
    int n = b * 256 + j;
    float v = alpha[n] * h[(size_t)n * 128 + c];
    mx = fmaxf(mx, v);
  }
  hpool[b * 128 + c] = lrelu_f(mx, 0.01f);
}

// bfuse[c] = sum_k aU2.b[k] * W1[150+k, c]
__global__ void fuse_bias(const float* __restrict__ aU2b, const float* __restrict__ W1,
                          float* bf) {
  int c = blockIdx.x * blockDim.x + threadIdx.x;
  if (c >= 150) return;
  float s = 0.f;
  for (int k = 0; k < 150; ++k) s += aU2b[k] * W1[(150 + k) * 150 + c];
  bf[c] = s;
}

// ei[n] = tanh(g1[batch] + g2[n] + bfuse + b1) . W2 + b2   (one wave / node)
// g1/g2 have padded ld 160; only the first 150 channels are used.
__global__ void attn_ei(const float* __restrict__ g1, const float* __restrict__ g2,
                        const float* __restrict__ bf, const float* __restrict__ b1,
                        const float* __restrict__ w2, const float* __restrict__ b2,
                        const int* __restrict__ batch, float* ei_out, int NT) {
  int warp = threadIdx.x >> 5, lane = threadIdx.x & 31;
  int n = blockIdx.x * 4 + warp;
  if (n >= NT) return;
  int b = batch[n];
  float p = 0.f;
  for (int c = lane; c < 150; c += 32) {
    float t = tanhf(g1[b * 160 + c] + g2[(size_t)n * 160 + c] + bf[c] + b1[c]);
    p += t * w2[c];
  }
  for (int off = 16; off; off >>= 1) p += __shfl_down(p, off, 32);
  if (lane == 0) ei_out[n] = p + b2[0];
}

// segment softmax over contiguous 256-node segments (LDS reductions)
__global__ void seg_softmax256(const float* __restrict__ ei, float* alpha) {
  __shared__ float sm[256];
  int b = blockIdx.x, t = threadIdx.x;
  float v = ei[b * 256 + t];
  sm[t] = v; __syncthreads();
  for (int off = 128; off; off >>= 1) { if (t < off) sm[t] = fmaxf(sm[t], sm[t + off]); __syncthreads(); }
  float m = sm[0]; __syncthreads();
  float e = expf(v - m);
  sm[t] = e; __syncthreads();
  for (int off = 128; off; off >>= 1) { if (t < off) sm[t] += sm[t + off]; __syncthreads(); }
  alpha[b * 256 + t] = e / (sm[0] + 1e-16f);
}

// xcat[256,1792] = [x_prot(512) | x_smile(500, ld 512) | prot_esm(768) | 0 pad]
__global__ void concat3(const float* __restrict__ xp, const float* __restrict__ xs,
                        const float* __restrict__ esm, float* xc, int total) {
  int i = blockIdx.x * blockDim.x + threadIdx.x;
  if (i >= total) return;
  int b = i / 1792, c = i % 1792;
  float v = (c < 512)  ? xp[b * 512 + c]
          : (c < 1012) ? xs[b * 512 + (c - 512)]
          : (c < 1780) ? esm[b * 768 + (c - 1012)]
                       : 0.f;
  xc[i] = v;
}

// ------------------------------ host helpers -------------------------------
static inline void gemm(hipStream_t st, const float* A, const float* B, const float* bias,
                        float* C, int M, int N, int K, int lda, int ldb, int ldc, int flags) {
  dim3 g((N + 15) / 16, (M + 15) / 16);
  gemm_wmma<<<g, 32, 0, st>>>(A, B, bias, C, M, N, K, lda, ldb, ldc, flags, 0.01f);
}

static inline void pad(hipStream_t st, const float* src, float* dst,
                       int rows, int cols, int lds, int rowsPad, int ldd) {
  int total = rowsPad * ldd;
  pad2d<<<(total + 255) / 256, 256, 0, st>>>(src, dst, rows, cols, lds, rowsPad, ldd);
}

struct GatPar { const float *W, *as, *ad, *Wr, *b, *We, *ae; };

// x has lda == Kpad (zero-padded cols); W/Wr have Kpad rows x 128 cols
static void run_gat(hipStream_t st, const float* x, int Kpad, int Nn,
                    const int* eidx, int E, const float* eattr, int fe,
                    const float* W, const float* Wr, const GatPar& p,
                    float* hbuf, float* outbuf,
                    float* elog, unsigned* nm, float* ns, float* dsrc, float* ddst,
                    float* wev, int do_lrelu) {
  gemm(st, x, W,  nullptr, hbuf,   Nn, 128, Kpad, Kpad, 128, 128, 0); // h = x @ W
  gemm(st, x, Wr, p.b,     outbuf, Nn, 128, Kpad, Kpad, 128, 128, 1); // x@Wres + bias
  node_dots<<<Nn, 128, 0, st>>>(hbuf, p.as, p.ad, dsrc, ddst);
  if (eattr) edge_wvec<<<1, 16, 0, st>>>(p.We, p.ae, wev);
  fill_u32<<<(Nn + 255) / 256, 256, 0, st>>>(nm, 0u, Nn);
  fill_f32<<<(Nn + 255) / 256, 256, 0, st>>>(ns, 0.f, Nn);
  edge_logit<<<(E + 255) / 256, 256, 0, st>>>(eidx, E, dsrc, ddst, eattr, wev, fe, elog, nm);
  edge_exp  <<<(E + 255) / 256, 256, 0, st>>>(eidx, E, elog, nm, ns);
  gat_scatter<<<E, 128, 0, st>>>(hbuf, elog, ns, eidx, E, outbuf);
  if (do_lrelu) lrelu_ip<<<((size_t)Nn * 128 + 255) / 256, 256, 0, st>>>(outbuf, Nn * 128, 0.01f);
}

extern "C" void kernel_launch(void* const* d_in, const int* in_sizes, int n_in,
                              void* d_out, int out_size, void* d_ws, size_t ws_size,
                              hipStream_t stream) {
  (void)in_sizes; (void)n_in; (void)out_size; (void)ws_size;
  const int NS = 8192, ES = 16384, NT = 65536, ET = 786432, B = 256;

  const float* x_s  = (const float*)d_in[0];
  const int*   ei_s = (const int*)  d_in[1];
  const float* ea_s = (const float*)d_in[2];
  /* x_s_batch d_in[3]: contiguous 32/graph (unused) */
  const float* x_t  = (const float*)d_in[4];
  const int*   ei_t = (const int*)  d_in[5];
  const int*   bt_t = (const int*)  d_in[6];
  const float* esm  = (const float*)d_in[7];

#define FP(i) ((const float*)d_in[(i)])
  int pi = 8;
  GatPar sg[4];
  for (int l = 0; l < 4; ++l) { sg[l] = { FP(pi), FP(pi+1), FP(pi+2), FP(pi+3), FP(pi+4), FP(pi+5), FP(pi+6) }; pi += 7; }
  const float* soutW = FP(pi++); const float* soutB = FP(pi++);
  GatPar pg[4];
  for (int l = 0; l < 4; ++l) { pg[l] = { FP(pi), FP(pi+1), FP(pi+2), FP(pi+3), FP(pi+4), nullptr, nullptr }; pi += 5; }
  const float* poutW = FP(pi++); const float* poutB = FP(pi++);
  const float* aU1W = FP(pi++), *aU1b = FP(pi++);
  const float* aU2W = FP(pi++), *aU2b = FP(pi++);
  const float* aW1W = FP(pi++), *aW1b = FP(pi++);
  const float* aW2W = FP(pi++), *aW2b = FP(pi++);
  const float* l1W  = FP(pi++), *l1b  = FP(pi++);
  const float* l2W  = FP(pi++), *l2b  = FP(pi++);
  const float* l3W  = FP(pi++), *l3b  = FP(pi++);
  const float* loW  = FP(pi++), *lob  = FP(pi++);
#undef FP

  // -------------------- workspace arena (~165 MB, 64B-aligned) ------------
  float* w = (float*)d_ws;
  size_t o = 0;
  auto alloc = [&](size_t n) { n = (n + 15) & ~(size_t)15; float* p = w + o; o += n; return p; };
  float*    ht_a   = alloc((size_t)NT * 128);
  float*    ht_b   = alloc((size_t)NT * 128);
  float*    big    = alloc((size_t)NT * 160);  // g2 (ld 160), then protein-GAT h
  float*    hs_a   = alloc((size_t)NS * 128);
  float*    hs_b   = alloc((size_t)NS * 128);
  float*    hs_h   = alloc((size_t)NS * 128);
  float*    elog   = alloc(ET);
  unsigned* nm     = (unsigned*)alloc(NT);
  float*    ns     = alloc(NT);
  float*    dsrc   = alloc(NT);
  float*    ddst   = alloc(NT);
  float*    wev    = alloc(16);
  float*    pooled = alloc((size_t)B * 256);
  float*    xsm    = alloc((size_t)B * 512);   // x_smile, ld 512 (cols 500.. = 0)
  float*    u1     = alloc((size_t)B * 160);
  float*    g1     = alloc((size_t)B * 160);
  float*    wfuse  = alloc((size_t)96 * 160);
  float*    bfuse  = alloc(160);
  float*    eibuf  = alloc(NT);
  float*    hpool  = alloc((size_t)B * 128);
  float*    xprot  = alloc((size_t)B * 512);
  float*    xcat   = alloc((size_t)B * 1792);
  float*    h1     = alloc((size_t)B * 1024);
  float*    h2     = alloc((size_t)B * 1024);
  float*    h3     = alloc((size_t)B * 512);
  // padded operand copies
  float*    xs_pad = alloc((size_t)NS * 64);   // x_s  [8192,44] -> ld 64
  float*    xt_pad = alloc((size_t)NT * 96);   // x_t  [65536,93] -> ld 96
  float*    sgW0p  = alloc((size_t)64 * 128);
  float*    sgWr0p = alloc((size_t)64 * 128);
  float*    pgW0p  = alloc((size_t)96 * 128);
  float*    pgWr0p = alloc((size_t)96 * 128);
  float*    soutWp = alloc((size_t)256 * 512); // [256,500] -> [256,512]
  float*    soutBp = alloc(512);
  float*    aU1Wp  = alloc((size_t)512 * 160); // [500,150] -> [512,160]
  float*    aU1bp  = alloc(160);
  float*    W1ap   = alloc((size_t)160 * 160); // aW1W rows 0..149
  float*    W1bp   = alloc((size_t)160 * 160); // aW1W rows 150..299
  float*    aU2Wp  = alloc((size_t)96 * 160);  // [93,150] -> [96,160]
  float*    l1Wp   = alloc((size_t)1792 * 1024);
  float*    loWp   = alloc((size_t)512 * 16);

  float* out_aff = (float*)d_out;       // [256, 1]
  float* alpha   = out_aff + 256;       // [256, 256] flat

  // -------------------- one-time zero-padding copies ----------------------
  pad(stream, x_s,   xs_pad, NS, 44, 44, NS, 64);
  pad(stream, x_t,   xt_pad, NT, 93, 93, NT, 96);
  pad(stream, sg[0].W,  sgW0p,  44, 128, 128, 64, 128);
  pad(stream, sg[0].Wr, sgWr0p, 44, 128, 128, 64, 128);
  pad(stream, pg[0].W,  pgW0p,  93, 128, 128, 96, 128);
  pad(stream, pg[0].Wr, pgWr0p, 93, 128, 128, 96, 128);
  pad(stream, soutW, soutWp, 256, 500, 500, 256, 512);
  pad(stream, soutB, soutBp, 1, 500, 500, 1, 512);
  pad(stream, aU1W,  aU1Wp, 500, 150, 150, 512, 160);
  pad(stream, aU1b,  aU1bp, 1, 150, 150, 1, 160);
  pad(stream, aW1W,             W1ap, 150, 150, 150, 160, 160);
  pad(stream, aW1W + 150 * 150, W1bp, 150, 150, 150, 160, 160);
  pad(stream, aU2W,  aU2Wp, 93, 150, 150, 96, 160);
  pad(stream, l1W,   l1Wp, 1780, 1024, 1024, 1792, 1024);
  pad(stream, loW,   loWp, 512, 1, 1, 512, 16);

  // ------------------------- SMILES encoder -------------------------------
  const float* cur = xs_pad; int Kp = 64;
  float* outs[4] = { hs_a, hs_b, hs_a, hs_b };
  for (int l = 0; l < 4; ++l) {
    const float* Wl  = (l == 0) ? sgW0p  : sg[l].W;
    const float* Wrl = (l == 0) ? sgWr0p : sg[l].Wr;
    run_gat(stream, cur, Kp, NS, ei_s, ES, ea_s, 11, Wl, Wrl, sg[l],
            hs_h, outs[l], elog, nm, ns, dsrc, ddst, wev, l < 3);
    cur = outs[l]; Kp = 128;
  }
  pool_smiles<<<B, 128, 0, stream>>>(cur, pooled);
  gemm(stream, pooled, soutWp, soutBp, xsm, B, 512, 256, 256, 512, 512, 1);

  // ------------------------- cross-modal attention ------------------------
  gemm(stream, xsm, aU1Wp, aU1bp, u1, B, 160, 512, 512, 160, 160, 1);
  gemm(stream, u1, W1ap, nullptr, g1, B, 160, 160, 160, 160, 160, 0);
  gemm(stream, aU2Wp, W1bp, nullptr, wfuse, 96, 160, 160, 160, 160, 160, 0);
  fuse_bias<<<1, 160, 0, stream>>>(aU2b, aW1W, bfuse);
  gemm(stream, xt_pad, wfuse, nullptr, big /*g2*/, NT, 160, 96, 96, 160, 160, 0);
  attn_ei<<<NT / 4, 128, 0, stream>>>(g1, big, bfuse, aW1b, aW2W, aW2b, bt_t, eibuf, NT);
  seg_softmax256<<<B, 256, 0, stream>>>(eibuf, alpha);

  // ------------------------- protein encoder ------------------------------
  cur = xt_pad; Kp = 96;
  float* ht_h = big;                    // g2 no longer needed
  float* outt[4] = { ht_a, ht_b, ht_a, ht_b };
  for (int l = 0; l < 4; ++l) {
    const float* Wl  = (l == 0) ? pgW0p  : pg[l].W;
    const float* Wrl = (l == 0) ? pgWr0p : pg[l].Wr;
    run_gat(stream, cur, Kp, NT, ei_t, ET, nullptr, 0, Wl, Wrl, pg[l],
            ht_h, outt[l], elog, nm, ns, dsrc, ddst, wev, l < 3);
    cur = outt[l]; Kp = 128;
  }
  pool_prot<<<B, 128, 0, stream>>>(cur, alpha, hpool);
  gemm(stream, hpool, poutW, poutB, xprot, B, 512, 128, 128, 512, 512, 1);

  // ------------------------- MLP head -------------------------------------
  concat3<<<(B * 1792 + 255) / 256, 256, 0, stream>>>(xprot, xsm, esm, xcat, B * 1792);
  gemm(stream, xcat, l1Wp, l1b, h1, B, 1024, 1792, 1792, 1024, 1024, 3);
  gemm(stream, h1,   l2W,  l2b, h2, B, 1024, 1024, 1024, 1024, 1024, 3);
  gemm(stream, h2,   l3W,  l3b, h3, B, 512,  1024, 1024, 512,  512,  3);
  gemm(stream, h3,   loWp, lob, out_aff, B, 1, 512, 512, 16, 1, 1);
}
